// BayesianNN_33904471835194
// MI455X (gfx1250) — matrix operand order repaired and checked by reading, not compiled, not run on hardware
//
#include <hip/hip_runtime.h>
#include <math.h>

// ---------------------------------------------------------------------------
// BayesianNN attention + NEAT relaxation for MI455X (gfx1250, wave32, WMMA)
//   D=7686 in, H=512 hidden, O=8 out, N=8206 graph nodes, M=2048 memories
//   DP = 7744 : D padded to a multiple of 64 (LDS K-step)
// GEMMs: async global->LDS double-buffered pipeline, 8 waves/block,
//        block tile 128x64, each wave a 32x32 accumulator block (4 WMMA tiles)
// ---------------------------------------------------------------------------

#define D_IN 7686
#define HID  512
#define ONUM 8
#define NN   (D_IN + HID + ONUM)   // 8206
#define MM   2048
#define DP   7744                  // /16=484 /32=242 /64=121

#define BM   128                   // block tile M
#define BN   64                    // block tile N
#define KB   64                    // K elements staged per step
#define KBP  72                    // padded LDS row stride (bank de-phasing)

typedef __attribute__((ext_vector_type(16))) __bf16 v16bf;
typedef __attribute__((ext_vector_type(8)))  __bf16 v8bf;
typedef __attribute__((ext_vector_type(8)))  float  v8f;

// ---- WMMA fragment loaders (16x16x32 bf16, wave32; contiguous k order, same for A and B) ----------
// A (16x32, MxK): lane L holds row M=L&15; half h=L>>4 selects
//   K = h*16+0..15  -> two contiguous 16B loads.
__device__ __forceinline__ v16bf load_a_frag(const __bf16* rowPtr, int kb, int h) {
    const v8bf lo = *(const v8bf*)(rowPtr + kb + h * 16);
    const v8bf hi = *(const v8bf*)(rowPtr + kb + h * 16 + 8);
    v16bf a;
#pragma unroll
    for (int e = 0; e < 8; ++e) { a[e] = lo[e]; a[e + 8] = hi[e]; }
    return a;
}
// B (32x16, KxN): lane L holds col N=L&15; half h selects K = h*16+0..15
__device__ __forceinline__ v16bf load_b_frag(const __bf16* colPtr, int kb, int h) {
    return *(const v16bf*)(colPtr + kb + h * 16);
}

// LDS byte offset of a __shared__ pointer: flat LDS-aperture addresses carry
// the DS offset in the low 32 bits (ISA aperture rules), so truncate.
__device__ __forceinline__ unsigned lds_off(const void* p) {
    return (unsigned)(unsigned long long)p;
}
// Async 16B global -> LDS copy (ASYNCcnt-tracked, no VGPR round trip)
__device__ __forceinline__ void async_cp_b128(unsigned ldsDst, const __bf16* gSrc) {
    asm volatile("global_load_async_to_lds_b128 %0, %1, off"
                 :: "v"(ldsDst), "v"(gSrc) : "memory");
}
__device__ __forceinline__ void wait_async0() {
    asm volatile("s_wait_asynccnt 0x0" ::: "memory");
}

// ---- unified LDS-staged GEMM ----------------------------------------------
// C[M,N] = A[M,K] * B[N,K]^T   (A rows K-contig, B "columns" K-contig)
// mode 0: outB[row*outLd+col]   = bf16(acc + bias[col])        (q, k)
// mode 1: outB[col*outLd+row]   = bf16(acc + bias[col])        (v transposed)
// mode 2: outF[row*outN +col]   = acc * scale                  (scores)
// mode 3: atomicAdd(ctx[col], sum_rows(acc)/MM)                (context pool)
__global__ __launch_bounds__(256)
void gemm_lds(const __bf16* __restrict__ A, const __bf16* __restrict__ B,
              int ld, int kTotal,
              const float* __restrict__ bias, float scale,
              __bf16* __restrict__ outB, int outLd,
              float* __restrict__ outF, int outN,
              float* __restrict__ ctx, int mode) {
    __shared__ __bf16 As[2][BM * KBP];
    __shared__ __bf16 Bs[2][BN * KBP];

    const int tid  = threadIdx.x;
    const int lane = tid & 31, wave = tid >> 5;
    const int wm = wave & 3, wn = wave >> 2;       // 4 x 2 wave grid
    const int m = lane & 15, h = lane >> 4;
    const int blockN0 = blockIdx.x * BN;
    const int blockM0 = blockIdx.y * BM;
    const int kSteps  = kTotal / KB;

    // per-step staging: A = 1024 16B chunks (4/thread), B = 512 (2/thread)
    auto issueTile = [&](int step, int buf) {
        const int kb0 = step * KB;
#pragma unroll
        for (int i = 0; i < 4; ++i) {
            int c = tid + 256 * i;
            int row = c >> 3, ko = (c & 7) * 8;
            async_cp_b128(lds_off(&As[buf][row * KBP + ko]),
                          A + (size_t)(blockM0 + row) * ld + kb0 + ko);
        }
#pragma unroll
        for (int i = 0; i < 2; ++i) {
            int c = tid + 256 * i;
            int col = c >> 3, ko = (c & 7) * 8;
            async_cp_b128(lds_off(&Bs[buf][col * KBP + ko]),
                          B + (size_t)(blockN0 + col) * ld + kb0 + ko);
        }
    };

    v8f acc[2][2] = {};

    issueTile(0, 0);
    for (int t = 0; t < kSteps; ++t) {
        wait_async0();        // own tile-t copies done
        __syncthreads();      // everyone's copies done; prev compute finished
        if (t + 1 < kSteps) issueTile(t + 1, (t + 1) & 1);

        const __bf16* as = &As[t & 1][0];
        const __bf16* bs = &Bs[t & 1][0];
#pragma unroll
        for (int ks = 0; ks < 2; ++ks) {   // two 16x16x32 sub-steps per KB
            const int kb = ks * 32;
            v16bf af[2], bfr[2];
#pragma unroll
            for (int s = 0; s < 2; ++s) {
                af[s]  = load_a_frag(as + (wm * 32 + s * 16 + m) * KBP, kb, h);
                bfr[s] = load_b_frag(bs + (wn * 32 + s * 16 + m) * KBP, kb, h);
            }
#pragma unroll
            for (int sm = 0; sm < 2; ++sm)
#pragma unroll
                for (int sn = 0; sn < 2; ++sn)
                    acc[sm][sn] = __builtin_amdgcn_wmma_f32_16x16x32_bf16(
                        false, af[sm], false, bfr[sn], (short)0,
                        acc[sm][sn], false, false);
        }
    }

    const int colBase = blockN0 + wn * 32;
    const int rowBase = blockM0 + wm * 32;
    if (mode == 3) {
#pragma unroll
        for (int sn = 0; sn < 2; ++sn) {
            float s = 0.0f;
#pragma unroll
            for (int sm = 0; sm < 2; ++sm)
#pragma unroll
                for (int r = 0; r < 8; ++r) s += acc[sm][sn][r];
            atomicAdd(&ctx[colBase + sn * 16 + m], s * (1.0f / MM));
        }
    } else if (mode == 2) {
#pragma unroll
        for (int sm = 0; sm < 2; ++sm)
#pragma unroll
            for (int sn = 0; sn < 2; ++sn) {
                int col = colBase + sn * 16 + m;
#pragma unroll
                for (int r = 0; r < 8; ++r) {
                    int row = rowBase + sm * 16 + h * 8 + r;
                    outF[(size_t)row * outN + col] = acc[sm][sn][r] * scale;
                }
            }
    } else {
#pragma unroll
        for (int sm = 0; sm < 2; ++sm)
#pragma unroll
            for (int sn = 0; sn < 2; ++sn) {
                int col = colBase + sn * 16 + m;
                float bv = (col < D_IN) ? bias[col] : 0.0f;
#pragma unroll
                for (int r = 0; r < 8; ++r) {
                    int row = rowBase + sm * 16 + h * 8 + r;
                    float v = acc[sm][sn][r] + bv;
                    if (mode == 0) outB[(size_t)row * outLd + col] = (__bf16)v;
                    else           outB[(size_t)col * outLd + row] = (__bf16)v;
                }
            }
    }
}

// ---- utility kernels ------------------------------------------------------
__global__ void zero_f32(float* __restrict__ a, int n) {
    int i = blockIdx.x * 256 + threadIdx.x;
    if (i < n) a[i] = 0.0f;
}

__global__ void cvt_input(const float* __restrict__ X, __bf16* __restrict__ Xb) {
    size_t idx = (size_t)blockIdx.x * 256 + threadIdx.x;
    if (idx >= (size_t)MM * DP) return;
    int r = (int)(idx / DP), c = (int)(idx % DP);
    Xb[idx] = (c < D_IN) ? (__bf16)X[(size_t)r * D_IN + c] : (__bf16)0.0f;
}

// W [D_IN,D_IN] f32 -> Wt [DP,DP] bf16, Wt[j*DP+i]=W[i*D+j] (LDS 32x32 tiles)
__global__ void cvt_weightT(const float* __restrict__ W, __bf16* __restrict__ Wt) {
    __shared__ float tile[32][33];
    const int i0 = blockIdx.y * 32, j0 = blockIdx.x * 32;
    const int tx = threadIdx.x, ty = threadIdx.y;
#pragma unroll
    for (int k = 0; k < 32; k += 8) {
        int i = i0 + ty + k, j = j0 + tx;
        tile[ty + k][tx] = (i < D_IN && j < D_IN) ? W[(size_t)i * D_IN + j] : 0.0f;
    }
    __syncthreads();
#pragma unroll
    for (int k = 0; k < 32; k += 8) {
        int j = j0 + ty + k, i = i0 + tx;
        Wt[(size_t)j * DP + i] = (__bf16)tile[tx][ty + k];
    }
}

// ---- row softmax: scores f32 -> probabilities bf16 ------------------------
__global__ void softmax_row(const float* __restrict__ scores,
                            __bf16* __restrict__ attnP) {
    const int row = blockIdx.x, tid = threadIdx.x;
    __shared__ float red[256];
    const float* s = scores + (size_t)row * MM;

    float mx = -3.4e38f;
    for (int j = tid; j < MM; j += 256) mx = fmaxf(mx, s[j]);
    red[tid] = mx; __syncthreads();
    for (int st = 128; st > 0; st >>= 1) {
        if (tid < st) red[tid] = fmaxf(red[tid], red[tid + st]);
        __syncthreads();
    }
    mx = red[0]; __syncthreads();

    float sum = 0.0f;
    for (int j = tid; j < MM; j += 256) sum += __expf(s[j] - mx);
    red[tid] = sum; __syncthreads();
    for (int st = 128; st > 0; st >>= 1) {
        if (tid < st) red[tid] += red[tid + st];
        __syncthreads();
    }
    const float inv = 1.0f / red[0];
    for (int j = tid; j < MM; j += 256)
        attnP[(size_t)row * MM + j] = (__bf16)(__expf(s[j] - mx) * inv);
}

// ---- sweep 1: partial[j] += sum_{i in slice} ctx[i] * (mu+sigma*eps)[i,j] --
__global__ void relax1_partial(const float* __restrict__ ctx,
                               const float* __restrict__ wmu,
                               const float* __restrict__ wsig,
                               const float* __restrict__ epsw,
                               float* __restrict__ partial) {
    const int j = blockIdx.x * 256 + threadIdx.x;
    if (j >= NN) return;
    const int i0 = blockIdx.y * 241;
    int i1 = i0 + 241; if (i1 > D_IN) i1 = D_IN;   // only rows i<D contribute
    float acc = 0.0f;
    for (int i = i0; i < i1; ++i) {
        size_t o = (size_t)i * NN + j;
        __builtin_prefetch(wmu + o + (size_t)8 * NN, 0, 0);
        acc = fmaf(ctx[i], fmaf(wsig[o], epsw[o], wmu[o]), acc);
    }
    atomicAdd(&partial[j], acc);
}

__global__ void relax1_final(const float* __restrict__ partial,
                             const float* __restrict__ bmu,
                             const float* __restrict__ bsig,
                             const float* __restrict__ epsb,
                             const float* __restrict__ ctx,
                             float* __restrict__ vals) {
    const int j = blockIdx.x * 256 + threadIdx.x;
    if (j >= NN) return;
    float b = fmaf(bsig[j], epsb[j], bmu[j]);
    float t = tanhf(partial[j] + b);
    vals[j] = (j < D_IN) ? ctx[j] : t;   // re-clamp input nodes
}

// ---- sweep 2: only output nodes j in [D+H, N) are ever read ----------------
__global__ void sweep2_out(const float* __restrict__ vals,
                           const float* __restrict__ wmu,
                           const float* __restrict__ wsig,
                           const float* __restrict__ epsw,
                           const float* __restrict__ bmu,
                           const float* __restrict__ bsig,
                           const float* __restrict__ epsb,
                           float* __restrict__ out) {
    const int j = D_IN + HID + blockIdx.x;   // 8 blocks, one output each
    const int tid = threadIdx.x;
    __shared__ float red[256];
    float acc = 0.0f;
    for (int i = tid; i < NN; i += 256) {
        size_t o = (size_t)i * NN + j;
        acc = fmaf(vals[i], fmaf(wsig[o], epsw[o], wmu[o]), acc);
    }
    red[tid] = acc; __syncthreads();
    for (int st = 128; st > 0; st >>= 1) {
        if (tid < st) red[tid] += red[tid + st];
        __syncthreads();
    }
    if (tid == 0) {
        float b = fmaf(bsig[j], epsb[j], bmu[j]);
        float t = tanhf(red[0] + b);
        out[blockIdx.x] = 1.0f / (1.0f + __expf(-t));
    }
}

// ---------------------------------------------------------------------------
extern "C" void kernel_launch(void* const* d_in, const int* in_sizes, int n_in,
                              void* d_out, int out_size, void* d_ws, size_t ws_size,
                              hipStream_t stream) {
    const float* X    = (const float*)d_in[0];
    const float* Wq   = (const float*)d_in[1];
    const float* bq   = (const float*)d_in[2];
    const float* Wk   = (const float*)d_in[3];
    const float* bk   = (const float*)d_in[4];
    const float* Wv   = (const float*)d_in[5];
    const float* bv   = (const float*)d_in[6];
    const float* wmu  = (const float*)d_in[7];
    const float* wsig = (const float*)d_in[8];
    const float* bmu  = (const float*)d_in[9];
    const float* bsig = (const float*)d_in[10];
    const float* epsw = (const float*)d_in[11];
    const float* epsb = (const float*)d_in[12];
    float* out = (float*)d_out;

    char* ws = (char*)d_ws;
    size_t off = 0;
    auto alloc = [&](size_t bytes) -> char* {
        char* p = ws + off;
        off = (off + bytes + 255) & ~(size_t)255;
        return p;
    };
    __bf16* Xb   = (__bf16*)alloc((size_t)MM * DP * 2);  // bf16 input
    __bf16* Wt   = (__bf16*)alloc((size_t)DP * DP * 2);  // W^T staging (x3 reuse)
    __bf16* qb   = (__bf16*)alloc((size_t)MM * DP * 2);
    __bf16* kbuf = (__bf16*)alloc((size_t)MM * DP * 2);
    __bf16* vtb  = (__bf16*)alloc((size_t)DP * MM * 2);  // v^T
    float*  sc   = (float*) alloc((size_t)MM * MM * 4);
    __bf16* aP   = (__bf16*)alloc((size_t)MM * MM * 2);
    float*  ctx  = (float*) alloc((size_t)DP * 4);
    float*  part = (float*) alloc((size_t)NN * 4);
    float*  vals = (float*) alloc((size_t)NN * 4);

    const float iscale = 1.0f / sqrtf((float)D_IN);
    const dim3 tblk(32, 8);
    const dim3 tgrid(DP / 32, DP / 32);        // 242 x 242 transpose tiles
    const dim3 gQKV(DP / BN, MM / BM);         // 121 x 16
    const dim3 gSc(MM / BN, MM / BM);          // 32 x 16

    zero_f32<<<(DP + 255) / 256, 256, 0, stream>>>(ctx, DP);
    zero_f32<<<(NN + 255) / 256, 256, 0, stream>>>(part, NN);

    cvt_input<<<(int)(((size_t)MM * DP + 255) / 256), 256, 0, stream>>>(X, Xb);

    // Q = X@Wq + bq ; K = X@Wk + bk ; V = X@Wv + bv (stored transposed)
    cvt_weightT<<<tgrid, tblk, 0, stream>>>(Wq, Wt);
    gemm_lds<<<gQKV, 256, 0, stream>>>(Xb, Wt, DP, DP, bq, 0.0f,
                                       qb, DP, nullptr, 0, nullptr, 0);
    cvt_weightT<<<tgrid, tblk, 0, stream>>>(Wk, Wt);
    gemm_lds<<<gQKV, 256, 0, stream>>>(Xb, Wt, DP, DP, bk, 0.0f,
                                       kbuf, DP, nullptr, 0, nullptr, 0);
    cvt_weightT<<<tgrid, tblk, 0, stream>>>(Wv, Wt);
    gemm_lds<<<gQKV, 256, 0, stream>>>(Xb, Wt, DP, DP, bv, 0.0f,
                                       vtb, MM, nullptr, 0, nullptr, 1);

    // scores = (q @ k^T)/sqrt(D) ; softmax ; context = mean(attnP @ v)
    gemm_lds<<<gSc, 256, 0, stream>>>(qb, kbuf, DP, DP, nullptr, iscale,
                                      nullptr, 0, sc, MM, nullptr, 2);
    softmax_row<<<MM, 256, 0, stream>>>(sc, aP);
    gemm_lds<<<gQKV, 256, 0, stream>>>(aP, vtb, MM, MM, nullptr, 0.0f,
                                       nullptr, 0, nullptr, 0, ctx, 3);

    // Bayesian-sampled graph relaxation (sweep 2 only needs the 8 out cols)
    relax1_partial<<<dim3((NN + 255) / 256, 32), 256, 0, stream>>>(ctx, wmu, wsig, epsw, part);
    relax1_final<<<(NN + 255) / 256, 256, 0, stream>>>(part, bmu, bsig, epsb, ctx, vals);
    sweep2_out<<<ONUM, 256, 0, stream>>>(vals, wmu, wsig, epsw, bmu, bsig, epsb, out);
}
